// guwenBERT_LSTM_CRF_38603166056618
// MI455X (gfx1250) — compile-verified, matrix operand order checked by reading, not run on hardware
//
#include <hip/hip_runtime.h>

// ---------------------------------------------------------------------------
// guwenBERT (4-layer) + BiLSTM + CRF decode for gfx1250 (MI455X), wave32 WMMA.
// All dense math runs through v_wmma_f32_16x16x32_f16 (f16 in, f32 accum).
// ---------------------------------------------------------------------------

typedef _Float16 half_t;
typedef __attribute__((ext_vector_type(16))) half_t v16h;
typedef __attribute__((ext_vector_type(8)))  half_t h8;
typedef __attribute__((ext_vector_type(4)))  half_t h4;
typedef __attribute__((ext_vector_type(8)))  float  v8f;

union H16U { v16h v; h8 h[2]; };

__device__ __forceinline__ float sigmf(float x) { return 1.0f / (1.0f + expf(-x)); }

// ---------------------------------------------------------------------------
// Generic WMMA GEMM: out = act(A[MxK](f16) * B[KxN](f16) + bias + c0)
//   bt=0: B stored [K,N] row-major (ldb = row stride)
//   bt=1: B stored [N,K] row-major (weights used transposed, ldb = row stride)
//   c0  : optional f32 addend, addr = c0 + row*c0_stride + col
//   act : 0 = none, 1 = exact GELU
// Block: 256 threads = 8 waves; tile 128(M) x 64(N); K-step 32.
// Wave w computes rows (w>>1)*32..+31, cols (w&1)*32..+31:
//   2 A-frags x 2 B-frags -> 4 WMMAs per K-step (ds:wmma = 2:1).
// ---------------------------------------------------------------------------
__launch_bounds__(256)
__global__ void gemm_wmma_kernel(const half_t* __restrict__ A, int lda,
                                 const half_t* __restrict__ B, int ldb, int bt,
                                 const float* __restrict__ bias,
                                 const float* __restrict__ c0, long long c0_stride,
                                 float* __restrict__ outF, half_t* __restrict__ outH,
                                 int ldo, int M, int N, int K, int act)
{
  __shared__ __align__(16) half_t At[128][32];
  __shared__ __align__(16) half_t Bt[64][32];   // [n][k]

  const int tid  = threadIdx.x;
  const int wave = tid >> 5;
  const int lane = tid & 31;
  const int mbase = blockIdx.y * 128;
  const int nbase = blockIdx.x * 64;
  const int wm = (wave >> 1) * 32;   // 0,32,64,96
  const int wn = (wave & 1) * 32;    // 0,32
  const int r16 = lane & 15;
  const int kb  = (lane >> 4) * 8;
  const int rowoff = (lane >> 4) * 8;

  v8f acc[2][2] = {};

  for (int k0 = 0; k0 < K; k0 += 32) {
    __syncthreads();
    // ---- stage A tile (128x32): thread -> row tid>>1, 16 halves ----
    {
      const int r  = tid >> 1;
      const int c16 = (tid & 1) * 16;
      const int gr = mbase + r;
      h8 v0 = {}, v1 = {};
      if (gr < M) {
        const half_t* ap = A + (size_t)gr * lda + k0 + c16;
        v0 = *(const h8*)(ap);
        v1 = *(const h8*)(ap + 8);
        if (k0 + 32 < K) __builtin_prefetch(ap + 32, 0, 1);
      }
      *(h8*)&At[r][c16]     = v0;
      *(h8*)&At[r][c16 + 8] = v1;
    }
    // ---- stage B tile into Bt[n][k] ----
    if (bt) {
      const int n  = tid >> 2;
      const int c8 = (tid & 3) * 8;
      const int gn = nbase + n;
      h8 val = {};
      if (gn < N) {
        const half_t* bp = B + (size_t)gn * ldb + k0 + c8;
        val = *(const h8*)(bp);
        if (k0 + 32 < K) __builtin_prefetch(bp + 32, 0, 1);
      }
      *(h8*)&Bt[n][c8] = val;
    } else if (tid < 64) {
      // thread: 4 consecutive k rows x 8 consecutive n cols, register transpose
      const int kk = (tid >> 3) * 4;
      const int n8 = (tid & 7) * 8;
      if (nbase + n8 + 8 <= N) {
        h8 row[4];
        #pragma unroll
        for (int i = 0; i < 4; ++i)
          row[i] = *(const h8*)(B + (size_t)(k0 + kk + i) * ldb + nbase + n8);
        #pragma unroll
        for (int j = 0; j < 8; ++j) {
          h4 col = { row[0][j], row[1][j], row[2][j], row[3][j] };
          *(h4*)&Bt[n8 + j][kk] = col;   // ds_store_b64
        }
      } else {
        #pragma unroll
        for (int i = 0; i < 4; ++i)
          #pragma unroll
          for (int j = 0; j < 8; ++j) {
            const int gn = nbase + n8 + j;
            Bt[n8 + j][kk + i] = (gn < N) ? B[(size_t)(k0 + kk + i) * ldb + gn] : (half_t)0;
          }
      }
    }
    __syncthreads();

    // ---- fragments (ISA 7.12.2 16-bit A layout) ----
    H16U a0, a1, b0, b1;
    a0.h[0] = *(const h8*)&At[wm + r16][kb];
    a0.h[1] = *(const h8*)&At[wm + r16][kb + 16];
    a1.h[0] = *(const h8*)&At[wm + 16 + r16][kb];
    a1.h[1] = *(const h8*)&At[wm + 16 + r16][kb + 16];
    b0.h[0] = *(const h8*)&Bt[wn + r16][kb];
    b0.h[1] = *(const h8*)&Bt[wn + r16][kb + 16];
    b1.h[0] = *(const h8*)&Bt[wn + 16 + r16][kb];
    b1.h[1] = *(const h8*)&Bt[wn + 16 + r16][kb + 16];

    acc[0][0] = __builtin_amdgcn_wmma_f32_16x16x32_f16(false, a0.v, false, b0.v, (short)0, acc[0][0], false, false);
    acc[0][1] = __builtin_amdgcn_wmma_f32_16x16x32_f16(false, a0.v, false, b1.v, (short)0, acc[0][1], false, false);
    acc[1][0] = __builtin_amdgcn_wmma_f32_16x16x32_f16(false, a1.v, false, b0.v, (short)0, acc[1][0], false, false);
    acc[1][1] = __builtin_amdgcn_wmma_f32_16x16x32_f16(false, a1.v, false, b1.v, (short)0, acc[1][1], false, false);
  }

  // ---- epilogue (C layout: lane = N col, VGPR = M row, +8 rows for lanes>=16) --
  #pragma unroll
  for (int mi = 0; mi < 2; ++mi) {
    #pragma unroll
    for (int r = 0; r < 8; ++r) {
      const int grow = mbase + wm + mi * 16 + r + rowoff;
      if (grow >= M) continue;
      #pragma unroll
      for (int ni = 0; ni < 2; ++ni) {
        const int gcol = nbase + wn + ni * 16 + r16;
        if (gcol >= N) continue;
        float v = acc[mi][ni][r];
        if (bias) v += bias[gcol];
        if (c0)   v += c0[(size_t)grow * (size_t)c0_stride + gcol];
        if (act == 1) v = 0.5f * v * (1.0f + erff(v * 0.70710678118654752f));
        if (outF) outF[(size_t)grow * ldo + gcol] = v;
        if (outH) outH[(size_t)grow * ldo + gcol] = (half_t)v;
      }
    }
  }
}

// ---------------------------------------------------------------------------
// Fused attention: one wave per (batch, head, 16-row q-tile). L=256, DH=64.
// S = (Q Kt) * 0.125 + maskbias; softmax over keys; ctx = P V (all via WMMA).
// ---------------------------------------------------------------------------
__launch_bounds__(32)
__global__ void attn_kernel(const half_t* __restrict__ Q, const half_t* __restrict__ Kt,
                            const half_t* __restrict__ V, const int* __restrict__ lens,
                            half_t* __restrict__ ctx)
{
  __shared__ __align__(16) half_t pbuf[16][256];
  __shared__ __align__(16) half_t vtile[32][64];

  const int lane = threadIdx.x;
  const int bid  = blockIdx.x;
  const int qt = bid & 15;
  const int h  = (bid >> 4) & 15;
  const int b  = bid >> 8;
  const int len = lens[b];
  const int r16 = lane & 15;
  const int kb  = (lane >> 4) * 8;
  const int rowoff = (lane >> 4) * 8;

  H16U q0, q1;
  {
    const size_t qr = ((size_t)b * 256 + qt * 16 + r16) * 1024 + (size_t)h * 64;
    q0.h[0] = *(const h8*)(Q + qr + kb);
    q0.h[1] = *(const h8*)(Q + qr + kb + 16);
    q1.h[0] = *(const h8*)(Q + qr + 32 + kb);
    q1.h[1] = *(const h8*)(Q + qr + 32 + kb + 16);
  }

  v8f s[16];
  #pragma unroll
  for (int kt = 0; kt < 16; ++kt) {
    const size_t kr = ((size_t)b * 256 + kt * 16 + r16) * 1024 + (size_t)h * 64;
    H16U k0, k1;
    k0.h[0] = *(const h8*)(Kt + kr + kb);
    k0.h[1] = *(const h8*)(Kt + kr + kb + 16);
    k1.h[0] = *(const h8*)(Kt + kr + 32 + kb);
    k1.h[1] = *(const h8*)(Kt + kr + 32 + kb + 16);
    v8f acc = {};
    acc = __builtin_amdgcn_wmma_f32_16x16x32_f16(false, q0.v, false, k0.v, (short)0, acc, false, false);
    acc = __builtin_amdgcn_wmma_f32_16x16x32_f16(false, q1.v, false, k1.v, (short)0, acc, false, false);
    const int key = kt * 16 + r16;
    const float mb = (key < len) ? 0.0f : -1e9f;
    #pragma unroll
    for (int r = 0; r < 8; ++r) s[kt][r] = acc[r] * 0.125f + mb;
  }

  // row max / sum over keys (rows live in 16-lane halves; xor masks 1..8 stay inside)
  float rm[8], rs[8];
  #pragma unroll
  for (int r = 0; r < 8; ++r) rm[r] = -3.0e38f;
  #pragma unroll
  for (int kt = 0; kt < 16; ++kt)
    #pragma unroll
    for (int r = 0; r < 8; ++r) rm[r] = fmaxf(rm[r], s[kt][r]);
  #pragma unroll
  for (int m = 1; m <= 8; m <<= 1)
    #pragma unroll
    for (int r = 0; r < 8; ++r) rm[r] = fmaxf(rm[r], __shfl_xor(rm[r], m, 32));
  #pragma unroll
  for (int r = 0; r < 8; ++r) rs[r] = 0.0f;
  #pragma unroll
  for (int kt = 0; kt < 16; ++kt)
    #pragma unroll
    for (int r = 0; r < 8; ++r) { float p = expf(s[kt][r] - rm[r]); s[kt][r] = p; rs[r] += p; }
  #pragma unroll
  for (int m = 1; m <= 8; m <<= 1)
    #pragma unroll
    for (int r = 0; r < 8; ++r) rs[r] += __shfl_xor(rs[r], m, 32);

  // P -> LDS in row-major [qrow][key]
  #pragma unroll
  for (int kt = 0; kt < 16; ++kt)
    #pragma unroll
    for (int r = 0; r < 8; ++r) pbuf[r + rowoff][kt * 16 + r16] = (half_t)s[kt][r];
  __syncthreads();

  v8f o[4] = {};
  #pragma unroll
  for (int kt2 = 0; kt2 < 8; ++kt2) {
    __syncthreads();
    {
      const size_t vr = ((size_t)b * 256 + kt2 * 32 + lane) * 1024 + (size_t)h * 64;
      #pragma unroll
      for (int c8 = 0; c8 < 8; ++c8)
        *(h8*)&vtile[lane][c8 * 8] = *(const h8*)(V + vr + c8 * 8);
    }
    __syncthreads();
    H16U pa;
    pa.h[0] = *(const h8*)&pbuf[r16][kt2 * 32 + kb];
    pa.h[1] = *(const h8*)&pbuf[r16][kt2 * 32 + kb + 16];
    #pragma unroll
    for (int nt = 0; nt < 4; ++nt) {
      H16U vb;
      const int dh = nt * 16 + r16;
      #pragma unroll
      for (int j = 0; j < 8; ++j) {
        vb.v[j]     = vtile[kb + j][dh];
        vb.v[8 + j] = vtile[kb + 16 + j][dh];
      }
      o[nt] = __builtin_amdgcn_wmma_f32_16x16x32_f16(false, pa.v, false, vb.v, (short)0, o[nt], false, false);
    }
  }

  #pragma unroll
  for (int nt = 0; nt < 4; ++nt)
    #pragma unroll
    for (int r = 0; r < 8; ++r) {
      const int qrow = qt * 16 + r + rowoff;
      const size_t orow = ((size_t)b * 256 + qrow) * 1024 + (size_t)h * 64 + nt * 16 + r16;
      ctx[orow] = (half_t)(o[nt][r] / rs[r]);
    }
}

// ---------------------------------------------------------------------------
// LayerNorm over D=1024 with optional residual: out = LN(base + delta)*g + b
// ---------------------------------------------------------------------------
__launch_bounds__(256)
__global__ void resid_ln_kernel(const float* __restrict__ base, const float* __restrict__ delta,
                                const float* __restrict__ g, const float* __restrict__ bt,
                                float* __restrict__ o32, half_t* __restrict__ o16)
{
  __shared__ float red[256];
  const int row = blockIdx.x, tid = threadIdx.x;
  const float* pb = base + (size_t)row * 1024;
  const float* pd = delta ? delta + (size_t)row * 1024 : nullptr;
  float v[4];
  float s = 0.0f;
  #pragma unroll
  for (int i = 0; i < 4; ++i) {
    const int c = tid + i * 256;
    float x = pb[c] + (pd ? pd[c] : 0.0f);
    v[i] = x; s += x;
  }
  red[tid] = s; __syncthreads();
  for (int st = 128; st > 0; st >>= 1) { if (tid < st) red[tid] += red[tid + st]; __syncthreads(); }
  const float mean = red[0] * (1.0f / 1024.0f);
  __syncthreads();
  float s2 = 0.0f;
  #pragma unroll
  for (int i = 0; i < 4; ++i) { float d = v[i] - mean; s2 += d * d; }
  red[tid] = s2; __syncthreads();
  for (int st = 128; st > 0; st >>= 1) { if (tid < st) red[tid] += red[tid + st]; __syncthreads(); }
  const float rstd = rsqrtf(red[0] * (1.0f / 1024.0f) + 1e-12f);
  #pragma unroll
  for (int i = 0; i < 4; ++i) {
    const int c = tid + i * 256;
    const float y = (v[i] - mean) * rstd * g[c] + bt[c];
    o32[(size_t)row * 1024 + c] = y;
    if (o16) o16[(size_t)row * 1024 + c] = (half_t)y;
  }
}

// ---------------------------------------------------------------------------
__global__ void embed_kernel(const int* __restrict__ ids, const float* __restrict__ tok,
                             const float* __restrict__ pos, float* __restrict__ out)
{
  const long long idx = (long long)blockIdx.x * 256 + threadIdx.x;
  if (idx >= (long long)8192 * 1024) return;
  const int row = (int)(idx >> 10), c = (int)(idx & 1023);
  const int l = row & 255;
  out[idx] = tok[(size_t)ids[row] * 1024 + c] + pos[(size_t)l * 1024 + c];
}

__global__ void f32_to_f16_kernel(const float* __restrict__ s, half_t* __restrict__ d, long long n)
{
  long long i = (long long)blockIdx.x * 256 + threadIdx.x;
  const long long stride = (long long)gridDim.x * 256;
  for (; i < n; i += stride) d[i] = (half_t)s[i];
}

__global__ void add_vec_kernel(const float* __restrict__ a, const float* __restrict__ b,
                               float* __restrict__ o, int n)
{
  const int i = blockIdx.x * 256 + threadIdx.x;
  if (i < n) o[i] = a[i] + b[i];
}

__global__ void fill_f16_kernel(half_t* p, long long n)
{ const long long i = (long long)blockIdx.x * 256 + threadIdx.x; if (i < n) p[i] = (half_t)0; }

__global__ void fill_f32_kernel(float* p, long long n)
{ const long long i = (long long)blockIdx.x * 256 + threadIdx.x; if (i < n) p[i] = 0.0f; }

// ---------------------------------------------------------------------------
// LSTM pointwise update (PyTorch gate order i,f,g,o); writes f16 state + hcat.
// ---------------------------------------------------------------------------
__global__ void lstm_update_kernel(const float* __restrict__ G, float* __restrict__ C,
                                   half_t* __restrict__ Hst, half_t* __restrict__ hcat,
                                   int t, int dirOff)
{
  const int idx = blockIdx.x * 256 + threadIdx.x;
  if (idx >= 32 * 512) return;
  const int b = idx >> 9, j = idx & 511;
  const float* gb = G + (size_t)b * 2048;
  const float ig = sigmf(gb[j]);
  const float fg = sigmf(gb[512 + j]);
  const float gg = tanhf(gb[1024 + j]);
  const float og = sigmf(gb[1536 + j]);
  const float c = fg * C[idx] + ig * gg;
  C[idx] = c;
  const float hv = og * tanhf(c);
  Hst[idx] = (half_t)hv;
  hcat[((size_t)b * 256 + t) * 1024 + dirOff + j] = (half_t)hv;
}

// ---------------------------------------------------------------------------
// CRF Viterbi decode, one lane per batch row (B=32, T=9, L=256).
// ---------------------------------------------------------------------------
__launch_bounds__(32)
__global__ void crf_kernel(const float* __restrict__ em, const int* __restrict__ lens,
                           const float* __restrict__ start, const float* __restrict__ endv,
                           const float* __restrict__ trans, unsigned char* __restrict__ hist,
                           float* __restrict__ tags)
{
  __shared__ float tr[81];
  const int b = threadIdx.x;
  for (int i = b; i < 81; i += 32) tr[i] = trans[i];
  __syncthreads();
  const int len = lens[b];
  float score[9];
  #pragma unroll
  for (int j = 0; j < 9; ++j) score[j] = start[j] + em[(size_t)b * 256 * 9 + j];
  for (int t = 1; t < 256; ++t) {
    const float* et = em + ((size_t)b * 256 + t) * 9;
    float ns[9];
    #pragma unroll
    for (int j = 0; j < 9; ++j) {
      float best = -3.4e38f; int bi = 0;
      #pragma unroll
      for (int i = 0; i < 9; ++i) {
        const float c = score[i] + tr[i * 9 + j];
        if (c > best) { best = c; bi = i; }
      }
      ns[j] = best + et[j];
      hist[((size_t)(t - 1) * 32 + b) * 9 + j] = (unsigned char)bi;
    }
    const bool mt = t < len;
    #pragma unroll
    for (int j = 0; j < 9; ++j) score[j] = mt ? ns[j] : score[j];
  }
  int cur = 0; float best = -3.4e38f;
  #pragma unroll
  for (int j = 0; j < 9; ++j) { const float c = score[j] + endv[j]; if (c > best) { best = c; cur = j; } }
  tags[(size_t)b * 256 + 255] = (255 < len) ? (float)cur : 0.0f;
  for (int t = 254; t >= 0; --t) {
    if ((t + 1) < len) cur = hist[((size_t)t * 32 + b) * 9 + cur];
    tags[(size_t)b * 256 + t] = (t < len) ? (float)cur : 0.0f;
  }
}

// ---------------------------------------------------------------------------
extern "C" void kernel_launch(void* const* d_in, const int* in_sizes, int n_in,
                              void* d_out, int out_size, void* d_ws, size_t ws_size,
                              hipStream_t stream)
{
  const int B = 32, L = 256, D = 1024, NH = 16, F = 4096, NL = 4, T = 9;
  const int Rows = B * L;  // 8192
  (void)in_sizes; (void)n_in; (void)out_size; (void)ws_size;

  const int*   ids  = (const int*)d_in[0];
  const int*   lens = (const int*)d_in[1];
  const float* tok  = (const float*)d_in[2];
  const float* pos  = (const float*)d_in[3];
  const float* eg   = (const float*)d_in[4];
  const float* ebi  = (const float*)d_in[5];
  const float* wq   = (const float*)d_in[6];  const float* bq = (const float*)d_in[7];
  const float* wk   = (const float*)d_in[8];  const float* bk = (const float*)d_in[9];
  const float* wv   = (const float*)d_in[10]; const float* bv = (const float*)d_in[11];
  const float* wo   = (const float*)d_in[12]; const float* bo = (const float*)d_in[13];
  const float* ln1g = (const float*)d_in[14]; const float* ln1b = (const float*)d_in[15];
  const float* w1   = (const float*)d_in[16]; const float* b1 = (const float*)d_in[17];
  const float* w2   = (const float*)d_in[18]; const float* b2 = (const float*)d_in[19];
  const float* ln2g = (const float*)d_in[20]; const float* ln2b = (const float*)d_in[21];
  const float* wihf = (const float*)d_in[22]; const float* whhf = (const float*)d_in[23];
  const float* bihf = (const float*)d_in[24]; const float* bhhf = (const float*)d_in[25];
  const float* wihb = (const float*)d_in[26]; const float* whhb = (const float*)d_in[27];
  const float* bihb = (const float*)d_in[28]; const float* bhhb = (const float*)d_in[29];
  const float* fcw  = (const float*)d_in[30]; const float* fcb = (const float*)d_in[31];
  const float* cst  = (const float*)d_in[32]; const float* cen = (const float*)d_in[33];
  const float* ctr  = (const float*)d_in[34];

  // ---- bump allocator on d_ws ----
  char* wsb = (char*)d_ws;
  size_t off = 0;
  auto alloc = [&](size_t bytes) -> void* {
    void* p = wsb + off;
    off += (bytes + 255) & ~(size_t)255;
    return p;
  };

  const size_t DDn = (size_t)NL * D * D;
  const size_t DFn = (size_t)NL * D * F;
  half_t* wq16 = (half_t*)alloc(DDn * 2);
  half_t* wk16 = (half_t*)alloc(DDn * 2);
  half_t* wv16 = (half_t*)alloc(DDn * 2);
  half_t* wo16 = (half_t*)alloc(DDn * 2);
  half_t* w116 = (half_t*)alloc(DFn * 2);
  half_t* w216 = (half_t*)alloc(DFn * 2);
  half_t* wihf16 = (half_t*)alloc((size_t)2048 * 1024 * 2);
  half_t* wihb16 = (half_t*)alloc((size_t)2048 * 1024 * 2);
  half_t* whhf16 = (half_t*)alloc((size_t)2048 * 512 * 2);
  half_t* whhb16 = (half_t*)alloc((size_t)2048 * 512 * 2);
  half_t* fc16   = (half_t*)alloc((size_t)1024 * 9 * 2);
  float*  biasF  = (float*)alloc(2048 * 4);
  float*  biasB  = (float*)alloc(2048 * 4);

  float*  X32   = (float*)alloc((size_t)Rows * D * 4);
  float*  Y32   = (float*)alloc((size_t)Rows * D * 4);
  float*  TMP32 = (float*)alloc((size_t)Rows * D * 4);
  half_t* X16   = (half_t*)alloc((size_t)Rows * D * 2);
  half_t* Y16   = (half_t*)alloc((size_t)Rows * D * 2);
  half_t* QKVC  = (half_t*)alloc((size_t)Rows * D * 2 * 4);  // Q,K,V,CTX contiguous
  half_t* Q16 = QKVC;
  half_t* K16 = Q16 + (size_t)Rows * D;
  half_t* V16 = K16 + (size_t)Rows * D;
  half_t* CTX16 = V16 + (size_t)Rows * D;
  half_t* FF16  = (half_t*)alloc((size_t)Rows * F * 2);
  float*  G32   = (float*)alloc((size_t)32 * 2048 * 4);
  half_t* Hst   = (half_t*)alloc((size_t)32 * 512 * 2);
  float*  Cst   = (float*)alloc((size_t)32 * 512 * 4);
  unsigned char* hist = (unsigned char*)alloc((size_t)255 * 32 * 9);

  // overlays (regions dead by the time they're reused)
  float*  PREF   = (float*)FF16;   // 8192*2048*4 == 8192*4096*2 bytes
  float*  PREB   = (float*)QKVC;   // 8192*2048*4 == 8192*1024*2*4 bytes
  half_t* HCAT16 = (half_t*)TMP32; // 16 MB <= 32 MB

  auto conv = [&](const float* s, half_t* d, long long n) {
    f32_to_f16_kernel<<<2048, 256, 0, stream>>>(s, d, n);
  };
  auto gemm = [&](const half_t* A, int lda, const half_t* Bm, int ldb, int bt,
                  const float* bias, const float* c0, long long c0s,
                  float* oF, half_t* oH, int ldo, int M, int N, int K, int act) {
    dim3 g((N + 63) / 64, (M + 127) / 128);
    gemm_wmma_kernel<<<g, 256, 0, stream>>>(A, lda, Bm, ldb, bt, bias, c0, c0s,
                                            oF, oH, ldo, M, N, K, act);
  };

  // ---- weight conversion to f16 ----
  conv(wq, wq16, (long long)DDn);
  conv(wk, wk16, (long long)DDn);
  conv(wv, wv16, (long long)DDn);
  conv(wo, wo16, (long long)DDn);
  conv(w1, w116, (long long)DFn);
  conv(w2, w216, (long long)DFn);
  conv(wihf, wihf16, (long long)2048 * 1024);
  conv(wihb, wihb16, (long long)2048 * 1024);
  conv(whhf, whhf16, (long long)2048 * 512);
  conv(whhb, whhb16, (long long)2048 * 512);
  conv(fcw, fc16, (long long)1024 * 9);
  add_vec_kernel<<<8, 256, 0, stream>>>(bihf, bhhf, biasF, 2048);
  add_vec_kernel<<<8, 256, 0, stream>>>(bihb, bhhb, biasB, 2048);

  // ---- embedding + LN ----
  embed_kernel<<<(Rows * D) / 256, 256, 0, stream>>>(ids, tok, pos, TMP32);
  resid_ln_kernel<<<Rows, 256, 0, stream>>>(TMP32, nullptr, eg, ebi, X32, X16);

  // ---- BERT layers ----
  for (int l = 0; l < NL; ++l) {
    const half_t* wql = wq16 + (size_t)l * D * D;
    const half_t* wkl = wk16 + (size_t)l * D * D;
    const half_t* wvl = wv16 + (size_t)l * D * D;
    const half_t* wol = wo16 + (size_t)l * D * D;
    gemm(X16, D, wql, D, 0, bq + (size_t)l * D, nullptr, 0, nullptr, Q16, D, Rows, D, D, 0);
    gemm(X16, D, wkl, D, 0, bk + (size_t)l * D, nullptr, 0, nullptr, K16, D, Rows, D, D, 0);
    gemm(X16, D, wvl, D, 0, bv + (size_t)l * D, nullptr, 0, nullptr, V16, D, Rows, D, D, 0);
    attn_kernel<<<B * NH * 16, 32, 0, stream>>>(Q16, K16, V16, lens, CTX16);
    gemm(CTX16, D, wol, D, 0, bo + (size_t)l * D, nullptr, 0, TMP32, nullptr, D, Rows, D, D, 0);
    resid_ln_kernel<<<Rows, 256, 0, stream>>>(X32, TMP32, ln1g + (size_t)l * D, ln1b + (size_t)l * D, Y32, Y16);
    gemm(Y16, D, w116 + (size_t)l * D * F, F, 0, b1 + (size_t)l * F, nullptr, 0,
         nullptr, FF16, F, Rows, F, D, 1);
    gemm(FF16, F, w216 + (size_t)l * F * D, D, 0, b2 + (size_t)l * D, nullptr, 0,
         TMP32, nullptr, D, Rows, D, F, 0);
    resid_ln_kernel<<<Rows, 256, 0, stream>>>(Y32, TMP32, ln2g + (size_t)l * D, ln2b + (size_t)l * D, X32, X16);
  }

  // ---- LSTM input projections (overlay regions now free) ----
  gemm(X16, D, wihf16, D, 1, biasF, nullptr, 0, PREF, nullptr, 2048, Rows, 2048, D, 0);
  gemm(X16, D, wihb16, D, 1, biasB, nullptr, 0, PREB, nullptr, 2048, Rows, 2048, D, 0);

  // ---- forward LSTM ----
  fill_f16_kernel<<<64, 256, 0, stream>>>(Hst, 32 * 512);
  fill_f32_kernel<<<64, 256, 0, stream>>>(Cst, 32 * 512);
  for (int t = 0; t < L; ++t) {
    gemm(Hst, 512, whhf16, 512, 1, nullptr, PREF + (size_t)t * 2048, (long long)L * 2048,
         G32, nullptr, 2048, B, 2048, 512, 0);
    lstm_update_kernel<<<64, 256, 0, stream>>>(G32, Cst, Hst, HCAT16, t, 0);
  }
  // ---- backward LSTM ----
  fill_f16_kernel<<<64, 256, 0, stream>>>(Hst, 32 * 512);
  fill_f32_kernel<<<64, 256, 0, stream>>>(Cst, 32 * 512);
  for (int t = L - 1; t >= 0; --t) {
    gemm(Hst, 512, whhb16, 512, 1, nullptr, PREB + (size_t)t * 2048, (long long)L * 2048,
         G32, nullptr, 2048, B, 2048, 512, 0);
    lstm_update_kernel<<<64, 256, 0, stream>>>(G32, Cst, Hst, HCAT16, t, 512);
  }

  // ---- emissions FC -> d_out, then CRF decode ----
  float* em = (float*)d_out;
  gemm(HCAT16, 1024, fc16, 9, 0, fcb, nullptr, 0, em, nullptr, T, Rows, T, 1024, 0);
  crf_kernel<<<1, 32, 0, stream>>>(em, lens, cst, cen, ctr, hist, em + (size_t)Rows * T);
}
